// BertREGraph_36309653520907
// MI455X (gfx1250) — matrix operand order repaired and checked by reading, not compiled, not run on hardware
//
#include <hip/hip_runtime.h>
#include <hip/hip_bf16.h>

// ---------------- constants ----------------
constexpr int B_  = 8;
constexpr int S_  = 512;
constexpr int H_  = 768;
constexpr int L_  = 12;
constexpr int NH_ = 12;
constexpr int DH_ = 64;          // H_/NH_
constexpr int FF_ = 3072;
constexpr int R_  = 8;
constexpr int E_  = 1024;
constexpr int YN_ = 6;
constexpr int RGH_ = 10;
constexpr long long BSH_ = (long long)B_ * S_ * H_;
constexpr long long BSF_ = (long long)B_ * S_ * FF_;

typedef __attribute__((ext_vector_type(16))) _Float16 v16h;
typedef __attribute__((ext_vector_type(8)))  _Float16 v8h;
typedef __attribute__((ext_vector_type(8)))  float    v8f;

union Frag { v16h v; v8h h[2]; };

// One k-step fragment set for a 32x64 wave tile: two A fragments + four B fragments.
struct FragSet { Frag a0, a1, b[4]; };

__device__ __forceinline__ void load_set(FragSet& f,
                                         const _Float16* __restrict__ arow0,
                                         const _Float16* __restrict__ arow1,
                                         const _Float16* __restrict__ brow0,
                                         const _Float16* __restrict__ brow1,
                                         const _Float16* __restrict__ brow2,
                                         const _Float16* __restrict__ brow3,
                                         int k, int hi)
{
    // A fragment: K-contiguous per lane half (matches B).
    // lanes 0-15 : halves0-15 = K k..k+15
    // lanes16-31 : halves0-15 = K k+16..k+31
    const _Float16* a0p = arow0 + k + hi * 16;
    const _Float16* a1p = arow1 + k + hi * 16;
    f.a0.h[0] = *(const v8h*)(a0p);
    f.a0.h[1] = *(const v8h*)(a0p + 8);
    f.a1.h[0] = *(const v8h*)(a1p);
    f.a1.h[1] = *(const v8h*)(a1p + 8);
    // B fragment: lanes 0-15 hold K k..k+15 of one column, lanes 16-31 K k+16..k+31
    // (K-contiguous in Bt).
    const _Float16* bp0 = brow0 + k + hi * 16;
    const _Float16* bp1 = brow1 + k + hi * 16;
    const _Float16* bp2 = brow2 + k + hi * 16;
    const _Float16* bp3 = brow3 + k + hi * 16;
    f.b[0].h[0] = *(const v8h*)(bp0); f.b[0].h[1] = *(const v8h*)(bp0 + 8);
    f.b[1].h[0] = *(const v8h*)(bp1); f.b[1].h[1] = *(const v8h*)(bp1 + 8);
    f.b[2].h[0] = *(const v8h*)(bp2); f.b[2].h[1] = *(const v8h*)(bp2 + 8);
    f.b[3].h[0] = *(const v8h*)(bp3); f.b[3].h[1] = *(const v8h*)(bp3 + 8);
}

__device__ __forceinline__ void mma_set(const FragSet& f, v8f acc0[4], v8f acc1[4])
{
    #pragma unroll
    for (int j = 0; j < 4; ++j) {
        acc0[j] = __builtin_amdgcn_wmma_f32_16x16x32_f16(
            false, f.a0.v, false, f.b[j].v, (short)0, acc0[j], false, false);
        acc1[j] = __builtin_amdgcn_wmma_f32_16x16x32_f16(
            false, f.a1.v, false, f.b[j].v, (short)0, acc1[j], false, false);
    }
}

// ---------------- WMMA GEMM ----------------
// C[M,N](f32) = A[M,K](f16, row-major, lda) * Bt[N,K](f16, K-contiguous, ldb) + bias[N]
// Batched over blockIdx.z with element strides. REQUIRES K % 64 == 0 (true for all
// shapes in this model: 768, 64, 512, 3072).
// Wave computes a 32x64 tile: 8 v_wmma_f32_16x16x32_f16 per k-step, register
// double-buffered with a SINGLE-EXIT pipelined loop (two k-steps per iteration,
// peeled tail) so accumulators stay tied in one register set (no phi copies) and
// next k-step's fragments load while current WMMAs execute.
// WCOLS==2: block = 8 waves as 4x2 -> 128x128 tile (N % 128 == 0, M % 128 == 0)
// WCOLS==1: block = 8 waves as 8x1 -> 256x64 tile  (N % 64 == 0,  M % 256 == 0)
// All guards compile-time so EXEC stays all-1s through every WMMA (ISA requirement).
template<int WCOLS, bool HAS_BIAS>
__launch_bounds__(256)
__global__ void k_wmma_gemm(const _Float16* __restrict__ A,  long long strideA, int lda,
                            const _Float16* __restrict__ Bt, long long strideB, int ldb,
                            float* __restrict__ C,           long long strideC, int ldc,
                            const float* __restrict__ bias, int K)
{
    const int lane = threadIdx.x & 31;
    const int wave = threadIdx.x >> 5;     // 0..7
    const int l16  = lane & 15;
    const int hi   = lane >> 4;            // 0 or 1 (lane half)

    const int waveRow = (WCOLS == 2) ? (wave >> 1) : wave;
    const int waveCol = (WCOLS == 2) ? (wave & 1) : 0;
    const int tileM = blockIdx.y * ((WCOLS == 2) ? 128 : 256) + waveRow * 32;
    const int tileN = blockIdx.x * (WCOLS * 64) + waveCol * 64;

    A  += (size_t)blockIdx.z * strideA;
    Bt += (size_t)blockIdx.z * strideB;
    C  += (size_t)blockIdx.z * strideC;

    const _Float16* arow0 = A + (size_t)(tileM + l16) * lda;
    const _Float16* arow1 = A + (size_t)(tileM + 16 + l16) * lda;
    const _Float16* brow0 = Bt + (size_t)(tileN +  0 + l16) * ldb;
    const _Float16* brow1 = Bt + (size_t)(tileN + 16 + l16) * ldb;
    const _Float16* brow2 = Bt + (size_t)(tileN + 32 + l16) * ldb;
    const _Float16* brow3 = Bt + (size_t)(tileN + 48 + l16) * ldb;

    v8f acc0[4], acc1[4];
    #pragma unroll
    for (int j = 0; j < 4; ++j) {
        float bv = 0.0f;
        if constexpr (HAS_BIAS) bv = bias[tileN + j * 16 + l16];
        #pragma unroll
        for (int r = 0; r < 8; ++r) { acc0[j][r] = bv; acc1[j][r] = bv; }
    }

    // Single-exit software-pipelined main loop; K is a multiple of 64 (>= 64).
    FragSet f0, f1;
    load_set(f0, arow0, arow1, brow0, brow1, brow2, brow3, 0, hi);
    int k = 32;
    while (k + 32 < K) {
        load_set(f1, arow0, arow1, brow0, brow1, brow2, brow3, k, hi);
        mma_set(f0, acc0, acc1);
        load_set(f0, arow0, arow1, brow0, brow1, brow2, brow3, k + 32, hi);
        mma_set(f1, acc0, acc1);
        k += 64;
    }
    // Tail: k == K - 32.
    load_set(f1, arow0, arow1, brow0, brow1, brow2, brow3, k, hi);
    mma_set(f0, acc0, acc1);
    mma_set(f1, acc0, acc1);

    // C/D layout: VGPR r -> row (+r, +8*hi), col = tileN + j*16 + l16
    #pragma unroll
    for (int j = 0; j < 4; ++j) {
        const int col = tileN + j * 16 + l16;
        #pragma unroll
        for (int r = 0; r < 8; ++r) {
            const int rr = tileM + r + hi * 8;
            C[(size_t)rr * ldc + col] = acc0[j][r];
            C[(size_t)(rr + 16) * ldc + col] = acc1[j][r];
        }
    }
}

// ---------------- helpers ----------------
__device__ inline float block_reduce_sum(float v, float* red) {
    int t = threadIdx.x;
    red[t] = v; __syncthreads();
    for (int s = 128; s > 0; s >>= 1) { if (t < s) red[t] += red[t + s]; __syncthreads(); }
    float r = red[0]; __syncthreads();
    return r;
}

// Embedding gather + LayerNorm. One block per token, 256 threads, H=768=3*256.
__launch_bounds__(256)
__global__ void k_embed(const int* __restrict__ ids,
                        const float* __restrict__ we, const float* __restrict__ pe,
                        const float* __restrict__ te,
                        const float* __restrict__ g, const float* __restrict__ be,
                        float* __restrict__ xo, _Float16* __restrict__ xo16)
{
    __shared__ float red[256];
    const int tok = blockIdx.x;
    const int s = tok % S_;
    const int id = ids[tok];
    float vals[3]; float sum = 0.f;
    #pragma unroll
    for (int i = 0; i < 3; ++i) {
        int c = threadIdx.x + i * 256;
        float x = we[(size_t)id * H_ + c] + pe[(size_t)s * H_ + c] + te[c];
        vals[i] = x; sum += x;
    }
    float mean = block_reduce_sum(sum, red) * (1.0f / H_);
    float vs = 0.f;
    #pragma unroll
    for (int i = 0; i < 3; ++i) { float d = vals[i] - mean; vs += d * d; }
    float var = block_reduce_sum(vs, red) * (1.0f / H_);
    float rstd = rsqrtf(var + 1e-12f);
    #pragma unroll
    for (int i = 0; i < 3; ++i) {
        int c = threadIdx.x + i * 256;
        float y = (vals[i] - mean) * rstd * g[c] + be[c];
        xo[(size_t)tok * H_ + c] = y;
        xo16[(size_t)tok * H_ + c] = (_Float16)y;
    }
}

// y = LN(x + c)*g + b ; in-place safe. One block per token.
__launch_bounds__(256)
__global__ void k_add_ln(const float* __restrict__ x, const float* __restrict__ cc,
                         const float* __restrict__ g, const float* __restrict__ be,
                         float* __restrict__ xo, _Float16* __restrict__ xo16)
{
    __shared__ float red[256];
    const int tok = blockIdx.x;
    float vals[3]; float sum = 0.f;
    #pragma unroll
    for (int i = 0; i < 3; ++i) {
        int c = threadIdx.x + i * 256;
        float v = x[(size_t)tok * H_ + c] + cc[(size_t)tok * H_ + c];
        vals[i] = v; sum += v;
    }
    float mean = block_reduce_sum(sum, red) * (1.0f / H_);
    float vs = 0.f;
    #pragma unroll
    for (int i = 0; i < 3; ++i) { float d = vals[i] - mean; vs += d * d; }
    float var = block_reduce_sum(vs, red) * (1.0f / H_);
    float rstd = rsqrtf(var + 1e-12f);
    #pragma unroll
    for (int i = 0; i < 3; ++i) {
        int c = threadIdx.x + i * 256;
        float y = (vals[i] - mean) * rstd * g[c] + be[c];
        xo[(size_t)tok * H_ + c] = y;
        xo16[(size_t)tok * H_ + c] = (_Float16)y;
    }
}

// Row softmax with pre-scale, f32 in -> f16 out. One block per row of S_ cols.
__launch_bounds__(256)
__global__ void k_softmax(const float* __restrict__ in, _Float16* __restrict__ out, float scale)
{
    __shared__ float red[256];
    const size_t base = (size_t)blockIdx.x * S_;
    const int t = threadIdx.x;
    float v0 = in[base + t] * scale;
    float v1 = in[base + t + 256] * scale;
    float m = fmaxf(v0, v1);
    red[t] = m; __syncthreads();
    for (int s = 128; s > 0; s >>= 1) { if (t < s) red[t] = fmaxf(red[t], red[t + s]); __syncthreads(); }
    float mx = red[0]; __syncthreads();
    float e0 = __expf(v0 - mx), e1 = __expf(v1 - mx);
    float sum = block_reduce_sum(e0 + e1, red);
    float inv = 1.0f / sum;
    out[base + t]       = (_Float16)(e0 * inv);
    out[base + t + 256] = (_Float16)(e1 * inv);
}

__global__ void k_f32_to_f16(const float* __restrict__ in, _Float16* __restrict__ out, long long n)
{
    long long t = (long long)blockIdx.x * blockDim.x + threadIdx.x;
    if (t < n) out[t] = (_Float16)in[t];
}

// Transposed weight convert: W[K,N] f32 -> Wt[N,K] f16
__global__ void k_wT(const float* __restrict__ W, _Float16* __restrict__ Wt, int K, int N)
{
    long long t = (long long)blockIdx.x * blockDim.x + threadIdx.x;
    if (t >= (long long)K * N) return;
    int n = (int)(t % N);
    int k = (int)(t / N);
    Wt[(size_t)n * K + k] = (_Float16)W[t];
}

// V [B,S,H] f32 -> Vt [B,NH,DH,S] f16
__global__ void k_vt(const float* __restrict__ v, _Float16* __restrict__ vt)
{
    long long t = (long long)blockIdx.x * blockDim.x + threadIdx.x;
    if (t >= BSH_) return;
    long long u = t;
    int s = (int)(u % S_); u /= S_;
    int d = (int)(u % DH_); u /= DH_;
    int h = (int)(u % NH_);
    int b = (int)(u / NH_);
    vt[t] = (_Float16)v[((size_t)b * S_ + s) * H_ + h * DH_ + d];
}

__global__ void k_gelu(const float* __restrict__ in, _Float16* __restrict__ out, long long n)
{
    long long t = (long long)blockIdx.x * blockDim.x + threadIdx.x;
    if (t < n) {
        float x = in[t];
        float y = 0.5f * x * (1.0f + erff(x * 0.70710678f));
        out[t] = (_Float16)y;
    }
}

__global__ void k_zero(float* __restrict__ p, long long n)
{
    long long t = (long long)blockIdx.x * blockDim.x + threadIdx.x;
    if (t < n) p[t] = 0.0f;
}

__global__ void k_axpy(float* __restrict__ x, const float* __restrict__ y, float a, long long n)
{
    long long t = (long long)blockIdx.x * blockDim.x + threadIdx.x;
    if (t < n) x[t] += a * y[t];
}

// RGCN edge messages: msg = h[src] @ Wr[et]; atomic scatter-add into agg[dst].
__global__ void k_rgcn_edge(const float* __restrict__ h,
                            const int* __restrict__ src, const int* __restrict__ dst,
                            const int* __restrict__ et,
                            const float* __restrict__ Wr,   // [R, di, dn]
                            float* __restrict__ agg,        // [B,S,768]
                            int di, int dn)
{
    long long t = (long long)blockIdx.x * blockDim.x + threadIdx.x;
    long long total = (long long)B_ * E_ * dn;
    if (t >= total) return;
    int o = (int)(t % dn);
    long long be = t / dn;
    int e = (int)(be % E_);
    int b = (int)(be / E_);
    int s = src[(size_t)b * E_ + e];
    int d = dst[(size_t)b * E_ + e];
    int r = et[(size_t)b * E_ + e];
    const float* hrow = h + ((size_t)b * S_ + s) * H_;
    const float* w = Wr + ((size_t)r * di) * dn + o;
    float acc = 0.f;
    for (int i = 0; i < di; ++i) acc += hrow[i] * w[(size_t)i * dn];
    atomicAdd(&agg[((size_t)b * S_ + d) * H_ + o], acc);
}

// h_out = agg + h @ Ws + b ; optional ReLU
__global__ void k_rgcn_self(const float* __restrict__ h, const float* __restrict__ Ws,
                            const float* __restrict__ bias, const float* __restrict__ agg,
                            float* __restrict__ out, int di, int dn, int relu)
{
    long long t = (long long)blockIdx.x * blockDim.x + threadIdx.x;
    long long total = (long long)B_ * S_ * dn;
    if (t >= total) return;
    int o = (int)(t % dn);
    long long bs = t / dn;
    const float* hrow = h + (size_t)bs * H_;
    float acc = bias[o] + agg[(size_t)bs * H_ + o];
    for (int i = 0; i < di; ++i) acc += hrow[i] * Ws[(size_t)i * dn + o];
    if (relu) acc = fmaxf(acc, 0.0f);
    out[(size_t)bs * H_ + o] = acc;
}

// att_logits[b,s] = enc[b,s,:] . att_w + att_b
__global__ void k_attlog(const float* __restrict__ enc, const float* __restrict__ w,
                         const float* __restrict__ bptr, float* __restrict__ out)
{
    int t = blockIdx.x * blockDim.x + threadIdx.x;
    if (t >= B_ * S_) return;
    const float* row = enc + (size_t)t * H_;
    float acc = bptr[0];
    for (int i = 0; i < H_; ++i) acc += row[i] * w[i];
    out[t] = acc;
}

// Span-pooled embedding. One block per batch element; sel picks feat half.
__launch_bounds__(256)
__global__ void k_span(const float* __restrict__ enc, const float* __restrict__ lg,
                       const int* __restrict__ spans, float* __restrict__ feat, int sel)
{
    __shared__ float w[S_];
    __shared__ float red[256];
    const int b = blockIdx.x;
    const int t = threadIdx.x;
    const int start = spans[b * 2 + 0];
    const int endv  = spans[b * 2 + 1];
    float lm = -3.0e38f;
    for (int j = t; j < S_; j += 256) {
        int idx = start + j;
        bool ok = (idx <= endv) && (idx < S_);
        int idxc = idx < (S_ - 1) ? idx : (S_ - 1);
        float val = ok ? lg[(size_t)b * S_ + idxc] : -1e30f;
        w[j] = val;
        lm = fmaxf(lm, val);
    }
    red[t] = lm; __syncthreads();
    for (int s = 128; s > 0; s >>= 1) { if (t < s) red[t] = fmaxf(red[t], red[t + s]); __syncthreads(); }
    float mx = red[0]; __syncthreads();
    float ls = 0.f;
    for (int j = t; j < S_; j += 256) {
        int idx = start + j;
        bool ok = (idx <= endv) && (idx < S_);
        float e = ok ? __expf(w[j] - mx) : 0.0f;
        w[j] = e; ls += e;
    }
    float sum = block_reduce_sum(ls, red);
    float inv = 1.0f / sum;
    for (int j = t; j < S_; j += 256) w[j] *= inv;
    __syncthreads();
    for (int d = t; d < H_; d += 256) {
        float acc = 0.f;
        for (int j = 0; j < S_; ++j) {
            int idx = start + j;
            int idxc = idx < (S_ - 1) ? idx : (S_ - 1);
            acc += w[j] * enc[((size_t)b * S_ + idxc) * H_ + d];
        }
        feat[((size_t)b * 2 + sel) * H_ + d] = acc;
    }
}

// logits = feat @ fc_w + fc_b ; log_softmax NLL mean loss. out[0..47]=logits, out[48]=loss
__launch_bounds__(64)
__global__ void k_fc_loss(const float* __restrict__ feat, const float* __restrict__ fc_w,
                          const float* __restrict__ fc_b, const int* __restrict__ data_y,
                          float* __restrict__ out)
{
    __shared__ float lg[B_ * YN_];
    const int t = threadIdx.x;
    if (t < B_ * YN_) {
        int b = t / YN_, y = t % YN_;
        float acc = fc_b[y];
        for (int k = 0; k < 2 * H_; ++k)
            acc += feat[(size_t)b * 2 * H_ + k] * fc_w[(size_t)k * YN_ + y];
        lg[t] = acc;
        out[t] = acc;
    }
    __syncthreads();
    if (t == 0) {
        float loss = 0.f, cnt = 0.f;
        for (int b = 0; b < B_; ++b) {
            int y = data_y[b];
            float valid = (y >= 0) ? 1.0f : 0.0f;
            int yy = y > 0 ? y : 0;
            float m = lg[b * YN_];
            for (int j = 1; j < YN_; ++j) m = fmaxf(m, lg[b * YN_ + j]);
            float s = 0.f;
            for (int j = 0; j < YN_; ++j) s += __expf(lg[b * YN_ + j] - m);
            float lp = lg[b * YN_ + yy] - m - __logf(s);
            loss += -lp * valid; cnt += valid;
        }
        out[B_ * YN_] = loss / fmaxf(cnt, 1.0f);
    }
}

// ---------------- host orchestration ----------------
static inline dim3 grid1(long long n) { return dim3((unsigned)((n + 255) / 256)); }

extern "C" void kernel_launch(void* const* d_in, const int* in_sizes, int n_in,
                              void* d_out, int out_size, void* d_ws, size_t ws_size,
                              hipStream_t stream)
{
    (void)in_sizes; (void)n_in; (void)out_size; (void)ws_size;

    const int* data_x    = (const int*)d_in[0];
    const int* data_e1   = (const int*)d_in[1];
    const int* data_e2   = (const int*)d_in[2];
    const int* data_y    = (const int*)d_in[3];
    const int* edge_src  = (const int*)d_in[4];
    const int* edge_dst  = (const int*)d_in[5];
    const int* edge_type = (const int*)d_in[6];

    const float* word_emb = (const float*)d_in[7];
    const float* pos_emb  = (const float*)d_in[8];
    const float* type_emb = (const float*)d_in[9];
    const float* emb_ln_s = (const float*)d_in[10];
    const float* emb_ln_b = (const float*)d_in[11];
    const float* Wq_all = (const float*)d_in[12];
    const float* bq_all = (const float*)d_in[13];
    const float* Wk_all = (const float*)d_in[14];
    const float* bk_all = (const float*)d_in[15];
    const float* Wv_all = (const float*)d_in[16];
    const float* bv_all = (const float*)d_in[17];
    const float* Wo_all = (const float*)d_in[18];
    const float* bo_all = (const float*)d_in[19];
    const float* Wi_all = (const float*)d_in[20];
    const float* bi_all = (const float*)d_in[21];
    const float* Wf_all = (const float*)d_in[22];
    const float* bf_all = (const float*)d_in[23];
    const float* ln1s_all = (const float*)d_in[24];
    const float* ln2s_all = (const float*)d_in[25];
    const float* ln1b_all = (const float*)d_in[26];
    const float* ln2b_all = (const float*)d_in[27];
    // d_in[28..48]: rg{i}_rel, rg{i}_self, rg{i}_b for i=0..6
    const float* att_w = (const float*)d_in[49];
    const float* att_b = (const float*)d_in[50];
    const float* fc_w  = (const float*)d_in[51];
    const float* fc_b  = (const float*)d_in[52];

    // ---- workspace carve ----
    char* ws = (char*)d_ws;
    size_t off = 0;
    auto carve = [&](size_t bytes) -> void* {
        void* p = ws + off;
        off = (off + bytes + 255) & ~(size_t)255;
        return p;
    };
    float*     xf32  = (float*)    carve(BSH_ * 4);
    _Float16*  xf16  = (_Float16*) carve(BSH_ * 2);
    float*     qf32  = (float*)    carve(BSH_ * 4);
    float*     kf32  = (float*)    carve(BSH_ * 4);
    float*     vf32  = (float*)    carve(BSH_ * 4);
    _Float16*  qf16  = (_Float16*) carve(BSH_ * 2);
    _Float16*  kf16  = (_Float16*) carve(BSH_ * 2);
    _Float16*  vt16  = (_Float16*) carve(BSH_ * 2);
    float*     scores= (float*)    carve((size_t)NH_ * S_ * S_ * 4);  // per-b reuse
    _Float16*  p16   = (_Float16*) carve((size_t)NH_ * S_ * S_ * 2);  // per-b reuse
    float*     ctx   = (float*)    carve(BSH_ * 4);
    float*     tmp1  = (float*)    carve(BSH_ * 4);
    float*     ff1   = (float*)    carve(BSF_ * 4);
    _Float16*  ff16  = (_Float16*) carve(BSF_ * 2);
    _Float16*  wq16  = (_Float16*) carve((size_t)H_ * H_ * 2);
    _Float16*  wk16  = (_Float16*) carve((size_t)H_ * H_ * 2);
    _Float16*  wv16  = (_Float16*) carve((size_t)H_ * H_ * 2);
    _Float16*  wo16  = (_Float16*) carve((size_t)H_ * H_ * 2);
    _Float16*  wi16  = (_Float16*) carve((size_t)H_ * FF_ * 2);
    _Float16*  wf16  = (_Float16*) carve((size_t)H_ * FF_ * 2);
    float*     h0    = (float*)    carve(BSH_ * 4);
    float*     h1    = (float*)    carve(BSH_ * 4);
    float*     agg   = (float*)    carve(BSH_ * 4);
    float*     attlog= (float*)    carve((size_t)B_ * S_ * 4);
    float*     feat  = (float*)    carve((size_t)B_ * 2 * H_ * 4);

    const int BS = B_ * S_;

    // ---- embeddings + LN ----
    k_embed<<<BS, 256, 0, stream>>>(data_x, word_emb, pos_emb, type_emb,
                                    emb_ln_s, emb_ln_b, xf32, xf16);

    // ---- BERT layers ----
    for (int l = 0; l < L_; ++l) {
        const float* Wq = Wq_all + (size_t)l * H_ * H_;
        const float* Wk = Wk_all + (size_t)l * H_ * H_;
        const float* Wv = Wv_all + (size_t)l * H_ * H_;
        const float* Wo = Wo_all + (size_t)l * H_ * H_;
        const float* Wi = Wi_all + (size_t)l * H_ * FF_;
        const float* Wf = Wf_all + (size_t)l * FF_ * H_;

        k_wT<<<grid1((long long)H_ * H_), 256, 0, stream>>>(Wq, wq16, H_, H_);
        k_wT<<<grid1((long long)H_ * H_), 256, 0, stream>>>(Wk, wk16, H_, H_);
        k_wT<<<grid1((long long)H_ * H_), 256, 0, stream>>>(Wv, wv16, H_, H_);
        k_wT<<<grid1((long long)H_ * H_), 256, 0, stream>>>(Wo, wo16, H_, H_);
        k_wT<<<grid1((long long)H_ * FF_), 256, 0, stream>>>(Wi, wi16, H_, FF_);
        k_wT<<<grid1((long long)H_ * FF_), 256, 0, stream>>>(Wf, wf16, FF_, H_);

        // QKV projections: [4096,768] x [768,768]
        dim3 gqkv(H_ / 128, BS / 128, 1);
        k_wmma_gemm<2, true><<<gqkv, 256, 0, stream>>>(
            xf16, 0, H_, wq16, 0, H_, qf32, 0, H_, bq_all + (size_t)l * H_, H_);
        k_wmma_gemm<2, true><<<gqkv, 256, 0, stream>>>(
            xf16, 0, H_, wk16, 0, H_, kf32, 0, H_, bk_all + (size_t)l * H_, H_);
        k_wmma_gemm<2, true><<<gqkv, 256, 0, stream>>>(
            xf16, 0, H_, wv16, 0, H_, vf32, 0, H_, bv_all + (size_t)l * H_, H_);

        k_f32_to_f16<<<grid1(BSH_), 256, 0, stream>>>(qf32, qf16, BSH_);
        k_f32_to_f16<<<grid1(BSH_), 256, 0, stream>>>(kf32, kf16, BSH_);
        k_vt<<<grid1(BSH_), 256, 0, stream>>>(vf32, vt16);

        // attention per batch element (scores buffer [NH,S,S] reused)
        for (int b = 0; b < B_; ++b) {
            dim3 gs(S_ / 128, S_ / 128, NH_);  // M=N=512, K=64, batch over heads
            k_wmma_gemm<2, false><<<gs, 256, 0, stream>>>(
                qf16 + (size_t)b * S_ * H_, (long long)DH_, H_,
                kf16 + (size_t)b * S_ * H_, (long long)DH_, H_,
                scores, (long long)S_ * S_, S_,
                nullptr, DH_);
            k_softmax<<<NH_ * S_, 256, 0, stream>>>(scores, p16, 0.125f);
            dim3 gc(DH_ / 64, S_ / 256, NH_);  // M=512, N=64, K=512 -> WCOLS=1
            k_wmma_gemm<1, false><<<gc, 256, 0, stream>>>(
                p16, (long long)S_ * S_, S_,
                vt16 + (size_t)b * S_ * H_, (long long)DH_ * S_, S_,
                ctx + (size_t)b * S_ * H_, (long long)DH_, H_,
                nullptr, S_);
        }

        // output projection + residual LN
        k_f32_to_f16<<<grid1(BSH_), 256, 0, stream>>>(ctx, qf16, BSH_);  // qf16 reused
        dim3 go(H_ / 128, BS / 128, 1);
        k_wmma_gemm<2, true><<<go, 256, 0, stream>>>(
            qf16, 0, H_, wo16, 0, H_, tmp1, 0, H_, bo_all + (size_t)l * H_, H_);
        k_add_ln<<<BS, 256, 0, stream>>>(xf32, tmp1,
                                         ln1s_all + (size_t)l * H_, ln1b_all + (size_t)l * H_,
                                         xf32, xf16);

        // FFN
        dim3 gi(FF_ / 128, BS / 128, 1);
        k_wmma_gemm<2, true><<<gi, 256, 0, stream>>>(
            xf16, 0, H_, wi16, 0, H_, ff1, 0, FF_, bi_all + (size_t)l * FF_, H_);
        k_gelu<<<grid1(BSF_), 256, 0, stream>>>(ff1, ff16, BSF_);
        dim3 gf(H_ / 128, BS / 128, 1);
        k_wmma_gemm<2, true><<<gf, 256, 0, stream>>>(
            ff16, 0, FF_, wf16, 0, FF_, tmp1, 0, H_, bf_all + (size_t)l * H_, FF_);
        k_add_ln<<<BS, 256, 0, stream>>>(xf32, tmp1,
                                         ln2s_all + (size_t)l * H_, ln2b_all + (size_t)l * H_,
                                         xf32, xf16);
    }

    // ---- RGCN ----
    hipMemcpyAsync(h0, xf32, (size_t)BSH_ * 4, hipMemcpyDeviceToDevice, stream);
    int dims[8] = { H_, RGH_, RGH_, RGH_, RGH_, RGH_, RGH_, H_ };
    float* ha = h0;
    float* hb = h1;
    for (int i = 0; i < 7; ++i) {
        int di = dims[i], dn = dims[i + 1];
        const float* Wr = (const float*)d_in[28 + 3 * i];
        const float* Wss = (const float*)d_in[29 + 3 * i];
        const float* bb = (const float*)d_in[30 + 3 * i];
        k_zero<<<grid1(BSH_), 256, 0, stream>>>(agg, BSH_);
        long long ne = (long long)B_ * E_ * dn;
        k_rgcn_edge<<<grid1(ne), 256, 0, stream>>>(ha, edge_src, edge_dst, edge_type,
                                                   Wr, agg, di, dn);
        long long ns = (long long)B_ * S_ * dn;
        k_rgcn_self<<<grid1(ns), 256, 0, stream>>>(ha, Wss, bb, agg, hb, di, dn,
                                                   (i < 6) ? 1 : 0);
        float* tsw = ha; ha = hb; hb = tsw;
    }
    k_axpy<<<grid1(BSH_), 256, 0, stream>>>(xf32, ha, 0.2f, BSH_);

    // ---- span attention head ----
    k_attlog<<<grid1((long long)BS), 256, 0, stream>>>(xf32, att_w, att_b, attlog);
    k_span<<<B_, 256, 0, stream>>>(xf32, attlog, data_e1, feat, 0);
    k_span<<<B_, 256, 0, stream>>>(xf32, attlog, data_e2, feat, 1);

    // ---- classifier + loss ----
    k_fc_loss<<<1, 64, 0, stream>>>(feat, fc_w, fc_b, data_y, (float*)d_out);
}
